// LinearTransformerLayer_44495861187342
// MI455X (gfx1250) — matrix operand order repaired and checked by reading, not compiled, not run on hardware
//
#include <hip/hip_runtime.h>
#include <math.h>

// ---------------------------------------------------------------------------
// Types for CDNA5 WMMA (wave32): v_wmma_f32_16x16x32_bf16
// ---------------------------------------------------------------------------
typedef __attribute__((ext_vector_type(16))) __bf16 v16bf;
typedef __attribute__((ext_vector_type(8)))  __bf16 bf16x8;
typedef __attribute__((ext_vector_type(8)))  float  v8f;

// ---------------------------------------------------------------------------
// Generic fused GEMM:  C[M,N] = act( scale * (A @ B(^T)) + bias )
//   A: [M,K] row-major, stride lda (f32)
//   TRANSB==0 -> B is [K,N] stride ldb ; TRANSB==1 -> B is [N,K] stride ldb
//   act: 0 none, 1 relu, 2 gelu(exact)
// Block: 256 thr = 8 waves, tile 128x128, wave tile 32x64 (2x4 WMMA),
// K-step 32, bf16 LDS staging, software-pipelined global->reg->LDS.
// ---------------------------------------------------------------------------
template <int TRANSB>
__global__ __launch_bounds__(256) void gemm_wmma_bf16(
    const float* __restrict__ A, int lda,
    const float* __restrict__ B, int ldb,
    const float* __restrict__ bias,
    float* __restrict__ C, int ldc,
    int M, int N, int K, float scale, int act)
{
    constexpr int BM = 128, BN = 128, BK = 32;
    constexpr int ASTR = 40, BSTR = 40;                 // padded bf16 strides (80B rows)
    __shared__ __align__(16) __bf16 As[BM * ASTR];
    __shared__ __align__(16) __bf16 Bs[BN * BSTR];      // stored as B^T: [n][k]

    const int tid  = threadIdx.x;
    const int lane = tid & 31;
    const int wave = tid >> 5;
    const int wr   = wave >> 1;      // 0..3  (M direction, 32 rows each)
    const int wc   = wave & 1;       // 0..1  (N direction, 64 cols each)
    const int hi   = (lane >> 4) & 1;
    const int l15  = lane & 15;

    const int bm = blockIdx.y * BM;
    const int bn = blockIdx.x * BN;

    // ---- A staging: thread -> (row, 16-wide k segment), contiguous global ----
    const int  ar   = tid >> 1;              // 0..127
    const int  aks  = (tid & 1) * 16;        // 0,16
    const bool avld = (bm + ar) < M;
    const float* Aptr = A + (size_t)(avld ? (bm + ar) : (M - 1)) * lda + aks;

    // ---- B staging ----
    // TRANSB==1: thread -> (n row of B, 16-wide k segment), contiguous global
    // TRANSB==0: thread -> (k row of B, 16-wide n segment), contiguous global,
    //            transposed on LDS write.
    const int  bn1  = tid >> 1;              // 0..127   (TRANSB==1)
    const int  bks1 = (tid & 1) * 16;
    const bool bvld = (bn + bn1) < N;
    const int  bkk0 = tid >> 3;              // 0..31    (TRANSB==0)
    const int  bns0 = (tid & 7) * 16;
    const float* Bptr = TRANSB
        ? B + (size_t)(bvld ? (bn + bn1) : (N > 0 ? N - 1 : 0)) * ldb + bks1
        : B + (size_t)bkk0 * ldb + bn + bns0;

    float areg[16], breg[16];

    auto loadA = [&](int k0) {
#pragma unroll
        for (int e = 0; e < 16; ++e) {
            float v = Aptr[k0 + e];
            areg[e] = avld ? v : 0.0f;
        }
    };
    auto loadB = [&](int k0) {
        if (TRANSB) {
#pragma unroll
            for (int e = 0; e < 16; ++e) {
                float v = Bptr[k0 + e];
                breg[e] = bvld ? v : 0.0f;
            }
        } else {
#pragma unroll
            for (int e = 0; e < 16; ++e) {
                float v = Bptr[(size_t)k0 * ldb + e];
                breg[e] = (bn + bns0 + e) < N ? v : 0.0f;
            }
        }
    };
    auto stage = [&]() {
        __bf16* ad = &As[ar * ASTR + aks];
#pragma unroll
        for (int e = 0; e < 16; ++e) ad[e] = (__bf16)areg[e];
        if (TRANSB) {
            __bf16* bd = &Bs[bn1 * BSTR + bks1];
#pragma unroll
            for (int e = 0; e < 16; ++e) bd[e] = (__bf16)breg[e];
        } else {
#pragma unroll
            for (int e = 0; e < 16; ++e) Bs[(bns0 + e) * BSTR + bkk0] = (__bf16)breg[e];
        }
    };

    v8f acc[2][4] = {};

    loadA(0); loadB(0);
    for (int k0 = 0; k0 < K; k0 += BK) {
        stage();
        __syncthreads();

        if (k0 + BK < K) {           // overlap next tile's global loads with WMMA
            loadA(k0 + BK); loadB(k0 + BK);
            if (k0 + 2 * BK < K) __builtin_prefetch(Aptr + k0 + 2 * BK, 0, 0);
        }

        // ---- fragments (ISA A/B VGPR layouts, wave32) ----
        // A 16x32: lanes0-15 row=l15 hold K0-7 & K16-23; lanes16-31 hold K8-15 & K24-31
        v16bf afr[2], bfr[4];
#pragma unroll
        for (int i = 0; i < 2; ++i) {
            const __bf16* p = &As[(wr * 32 + 16 * i + l15) * ASTR + hi * 8];
            bf16x8 lo = *reinterpret_cast<const bf16x8*>(p);
            bf16x8 ha = *reinterpret_cast<const bf16x8*>(p + 16);
#pragma unroll
            for (int e = 0; e < 8; ++e) { afr[i][e] = lo[e]; afr[i][e + 8] = ha[e]; }
        }
        // B 32x16 (B^T rows): lanes0-15 col=l15 hold K0-7 & K16-23; lanes16-31 hold K8-15 & K24-31
#pragma unroll
        for (int j = 0; j < 4; ++j) {
            const __bf16* p = &Bs[(wc * 64 + 16 * j + l15) * BSTR + hi * 8];
            bf16x8 lo = *reinterpret_cast<const bf16x8*>(p);
            bf16x8 ha = *reinterpret_cast<const bf16x8*>(p + 16);
#pragma unroll
            for (int e = 0; e < 8; ++e) { bfr[j][e] = lo[e]; bfr[j][e + 8] = ha[e]; }
        }

#pragma unroll
        for (int i = 0; i < 2; ++i)
#pragma unroll
            for (int j = 0; j < 4; ++j)
                acc[i][j] = __builtin_amdgcn_wmma_f32_16x16x32_bf16(
                    false, afr[i], false, bfr[j], (short)0, acc[i][j], false, false);

        __syncthreads();
    }

    // ---- epilogue: C layout VGPR r -> (M = base + r + 8*hi, N = base + l15) ----
#pragma unroll
    for (int i = 0; i < 2; ++i) {
#pragma unroll
        for (int j = 0; j < 4; ++j) {
            const int col = bn + wc * 64 + 16 * j + l15;
            if (col >= N) continue;
            const int rbase = bm + wr * 32 + 16 * i + 8 * hi;
            const float bv = bias ? bias[col] : 0.0f;
#pragma unroll
            for (int r = 0; r < 8; ++r) {
                const int row = rbase + r;
                if (row >= M) continue;
                float v = acc[i][j][r] * scale + bv;
                if (act == 1)      v = fmaxf(v, 0.0f);
                else if (act == 2) v = 0.5f * v * (1.0f + erff(v * 0.70710678118654752f));
                C[(size_t)row * ldc + col] = v;
            }
        }
    }
}

// ---------------------------------------------------------------------------
// In-place row softmax (one block per row)
// ---------------------------------------------------------------------------
__global__ __launch_bounds__(256) void softmax_rows_kernel(float* __restrict__ S, int ncols)
{
    float* p = S + (size_t)blockIdx.x * ncols;
    __shared__ float red[256];
    const int tid = threadIdx.x;

    float m = -3.4e38f;
    for (int c = tid; c < ncols; c += 256) m = fmaxf(m, p[c]);
    red[tid] = m; __syncthreads();
    for (int s = 128; s > 0; s >>= 1) { if (tid < s) red[tid] = fmaxf(red[tid], red[tid + s]); __syncthreads(); }
    m = red[0]; __syncthreads();

    float sum = 0.0f;
    for (int c = tid; c < ncols; c += 256) { float e = __expf(p[c] - m); p[c] = e; sum += e; }
    red[tid] = sum; __syncthreads();
    for (int s = 128; s > 0; s >>= 1) { if (tid < s) red[tid] += red[tid + s]; __syncthreads(); }
    const float inv = 1.0f / red[0];
    for (int c = tid; c < ncols; c += 256) p[c] *= inv;
}

// ---------------------------------------------------------------------------
// out = LayerNorm(X + Y) * g + b     (Y optional; one block per row, D cols)
// ---------------------------------------------------------------------------
__global__ __launch_bounds__(256) void layernorm_res_kernel(
    const float* __restrict__ X, const float* __restrict__ Y,
    const float* __restrict__ g, const float* __restrict__ b,
    float* __restrict__ out, int D)
{
    const size_t base = (size_t)blockIdx.x * D;
    __shared__ float s1[256], s2[256];
    const int tid = threadIdx.x;

    float sum = 0.0f, sq = 0.0f;
    for (int c = tid; c < D; c += 256) {
        float v = X[base + c] + (Y ? Y[base + c] : 0.0f);
        sum += v; sq += v * v;
    }
    s1[tid] = sum; s2[tid] = sq; __syncthreads();
    for (int s = 128; s > 0; s >>= 1) {
        if (tid < s) { s1[tid] += s1[tid + s]; s2[tid] += s2[tid + s]; }
        __syncthreads();
    }
    const float mu  = s1[0] / (float)D;
    const float var = s2[0] / (float)D - mu * mu;
    const float rstd = rsqrtf(var + 1e-5f);
    for (int c = tid; c < D; c += 256) {
        float v = X[base + c] + (Y ? Y[base + c] : 0.0f);
        out[base + c] = (v - mu) * rstd * g[c] + b[c];
    }
}

// fused = gate*a + (1-gate)*b
__global__ __launch_bounds__(256) void gate_fuse_kernel(
    const float* __restrict__ gate, const float* __restrict__ a,
    const float* __restrict__ b, float* __restrict__ o, size_t n)
{
    size_t i = (size_t)blockIdx.x * blockDim.x + threadIdx.x;
    if (i < n) { float gv = gate[0]; o[i] = gv * a[i] + (1.0f - gv) * b[i]; }
}

// ---------------------------------------------------------------------------
// Host orchestration
// ---------------------------------------------------------------------------
extern "C" void kernel_launch(void* const* d_in, const int* in_sizes, int n_in,
                              void* d_out, int out_size, void* d_ws, size_t ws_size,
                              hipStream_t stream)
{
    (void)in_sizes; (void)n_in; (void)out_size; (void)ws_size;
    const float* x      = (const float*)d_in[0];
    const float* mem    = (const float*)d_in[1];
    const float* sa_w   = (const float*)d_in[2];
    const float* sa_b   = (const float*)d_in[3];
    const float* sa_ow  = (const float*)d_in[4];
    const float* sa_ob  = (const float*)d_in[5];
    const float* ca_w   = (const float*)d_in[6];
    const float* ca_b   = (const float*)d_in[7];
    const float* ca_ow  = (const float*)d_in[8];
    const float* ca_ob  = (const float*)d_in[9];
    const float* f_w1   = (const float*)d_in[10];
    const float* f_b1   = (const float*)d_in[11];
    const float* f_w2   = (const float*)d_in[12];
    const float* f_b2   = (const float*)d_in[13];
    const float* ns_g   = (const float*)d_in[14];
    const float* ns_b   = (const float*)d_in[15];
    const float* nc_g   = (const float*)d_in[16];
    const float* nc_b   = (const float*)d_in[17];
    const float* nf_g   = (const float*)d_in[18];
    const float* nf_b   = (const float*)d_in[19];
    const float* gate   = (const float*)d_in[20];
    const float* ex_w   = (const float*)d_in[21];
    const float* ex_b   = (const float*)d_in[22];
    const float* sq_w   = (const float*)d_in[23];
    const float* sq_b   = (const float*)d_in[24];
    const float* nrm_g  = (const float*)d_in[25];
    const float* nrm_b  = (const float*)d_in[26];
    const float* q_w    = (const float*)d_in[27];
    const float* q_b    = (const float*)d_in[28];
    const float* k_w    = (const float*)d_in[29];
    const float* k_b    = (const float*)d_in[30];
    const float* v_w    = (const float*)d_in[31];
    const float* v_b    = (const float*)d_in[32];
    float* out = (float*)d_out;

    const int S = 4096, Dm = 1024;          // B*seq rows, model dim
    const size_t U = (size_t)S * Dm;        // 16.78 MB unit (floats)
    float* ws = (float*)d_ws;

    float* r0    = ws;            // 4U: sgira qkv(3U) -> ffn_mid(4U) -> saiga q|k|v|ctx
    float* ctx   = ws + 4 * U;    // 1U attention context (self & cross)
    float* selfs = ws + 5 * U;    // 1U
    float* caq   = ws + 6 * U;    // 1U
    float* cakv  = ws + 7 * U;    // 2U  (later: excite mid, N=2048)
    float* cross = ws + 9 * U;    // 1U
    float* fused = ws + 10 * U;   // 1U
    float* hid   = ws + 11 * U;   // 1U
    float* se    = ws + 12 * U;   // 1U
    float* tmp   = ws + 13 * U;   // 1U (out-proj / ffn-out / squeezed)
    float* score = ws + 14 * U;   // 1M floats, reused per head

    auto gemm = [&](const float* A, int lda, const float* B, int ldb, int tB,
                    const float* bias, float* C, int ldc,
                    int M, int N, int K, float scale, int act) {
        dim3 g((N + 127) / 128, (M + 127) / 128);
        if (tB)
            gemm_wmma_bf16<1><<<g, 256, 0, stream>>>(A, lda, B, ldb, bias, C, ldc,
                                                     M, N, K, scale, act);
        else
            gemm_wmma_bf16<0><<<g, 256, 0, stream>>>(A, lda, B, ldb, bias, C, ldc,
                                                     M, N, K, scale, act);
    };
    auto attn = [&](const float* Qb, int ldq, const float* Kb, int ldk,
                    const float* Vb, int ldv, float* Cb, int ldc_,
                    int nheads, int hd) {
        const float scl = rsqrtf((float)hd);
        for (int b = 0; b < 4; ++b) {
            for (int h = 0; h < nheads; ++h) {
                const float* Q = Qb + (size_t)b * 1024 * ldq + h * hd;
                const float* K = Kb + (size_t)b * 1024 * ldk + h * hd;
                const float* V = Vb + (size_t)b * 1024 * ldv + h * hd;
                float*       C = Cb + (size_t)b * 1024 * ldc_ + h * hd;
                gemm(Q, ldq, K, ldk, 1, nullptr, score, 1024, 1024, 1024, hd, scl, 0);
                softmax_rows_kernel<<<1024, 256, 0, stream>>>(score, 1024);
                gemm(score, 1024, V, ldv, 0, nullptr, C, ldc_, 1024, hd, 1024, 1.0f, 0);
            }
        }
    };

    // ---- SGIRA self-attention ----
    gemm(x, Dm, sa_w, 3 * Dm, 0, sa_b, r0, 3 * Dm, S, 3 * Dm, Dm, 1.0f, 0);   // qkv
    attn(r0, 3 * Dm, r0 + Dm, 3 * Dm, r0 + 2 * Dm, 3 * Dm, ctx, Dm, 16, 64);
    gemm(ctx, Dm, sa_ow, Dm, 0, sa_ob, tmp, Dm, S, Dm, Dm, 1.0f, 0);
    layernorm_res_kernel<<<S, 256, 0, stream>>>(x, tmp, ns_g, ns_b, selfs, Dm);

    // ---- SGIRA cross-attention ----
    gemm(selfs, Dm, ca_w, 3 * Dm, 0, ca_b, caq, Dm, S, Dm, Dm, 1.0f, 0);              // q
    gemm(mem, Dm, ca_w + Dm, 3 * Dm, 0, ca_b + Dm, cakv, 2 * Dm, S, 2 * Dm, Dm, 1.0f, 0); // k|v
    attn(caq, Dm, cakv, 2 * Dm, cakv + Dm, 2 * Dm, ctx, Dm, 16, 64);
    gemm(ctx, Dm, ca_ow, Dm, 0, ca_ob, tmp, Dm, S, Dm, Dm, 1.0f, 0);
    layernorm_res_kernel<<<S, 256, 0, stream>>>(selfs, tmp, nc_g, nc_b, cross, Dm);

    // ---- gate + FFN ----
    gate_fuse_kernel<<<(unsigned)((U + 255) / 256), 256, 0, stream>>>(gate, selfs, cross, fused, U);
    gemm(fused, Dm, f_w1, 4 * Dm, 0, f_b1, r0, 4 * Dm, S, 4 * Dm, Dm, 1.0f, 2);  // gelu
    gemm(r0, 4 * Dm, f_w2, Dm, 0, f_b2, tmp, Dm, S, Dm, 4 * Dm, 1.0f, 0);
    layernorm_res_kernel<<<S, 256, 0, stream>>>(fused, tmp, nf_g, nf_b, hid, Dm);

    // ---- SAIGA squeeze-excite ----
    gemm(hid, Dm, ex_w, 2 * Dm, 0, ex_b, cakv, 2 * Dm, S, 2 * Dm, Dm, 1.0f, 1);  // relu
    gemm(cakv, 2 * Dm, sq_w, Dm, 0, sq_b, tmp, Dm, S, Dm, 2 * Dm, 1.0f, 0);
    layernorm_res_kernel<<<S, 256, 0, stream>>>(hid, tmp, nrm_g, nrm_b, se, Dm);

    // ---- SAIGA attention (4 heads, hd=256) ----
    float* sq_ = r0;            // q
    float* sk_ = r0 + U;        // k
    float* sv_ = r0 + 2 * U;    // v
    float* sc_ = r0 + 3 * U;    // ctx
    gemm(se, Dm, q_w, Dm, 0, q_b, sq_, Dm, S, Dm, Dm, 1.0f, 0);
    gemm(se, Dm, k_w, Dm, 0, k_b, sk_, Dm, S, Dm, Dm, 1.0f, 0);
    gemm(se, Dm, v_w, Dm, 0, v_b, sv_, Dm, S, Dm, Dm, 1.0f, 0);
    attn(sq_, Dm, sk_, Dm, sv_, Dm, sc_, Dm, 4, 256);

    // ---- final norm -> output ----
    layernorm_res_kernel<<<S, 256, 0, stream>>>(se, sc_, nrm_g, nrm_b, out, Dm);
}
